// WindowAttention_83863531422077
// MI455X (gfx1250) — matrix operand order repaired and checked by reading, not compiled, not run on hardware
//
#include <hip/hip_runtime.h>

// ---------------------------------------------------------------------------
// Fused window attention for MI455X (gfx1250, wave32, WMMA 16x16x32 bf16).
// One 256-thread workgroup (8 waves) per window; everything resident in LDS.
// v3: fully packed f32->bf16 conversions (v_cvt_pk_bf16_f32) incl. all
//     epilogues, register-cached softmax exponentials with vector stores,
//     double-buffered QKV weight panel -> single barrier per k-step.
// ---------------------------------------------------------------------------

typedef __attribute__((ext_vector_type(16))) __bf16 v16bf;
typedef __attribute__((ext_vector_type(8)))  __bf16 v8bf;
typedef __attribute__((ext_vector_type(4)))  __bf16 v4bf;
typedef __attribute__((ext_vector_type(8)))  float  v8f;
typedef __attribute__((ext_vector_type(4)))  float  v4f;

#define SEQ   49      // tokens per window
#define NP    64      // padded tokens (4 tiles of 16)
#define CDIM  384
#define NH    12
#define HD    32

// LDS strides (elements); multiples of 8 -> 16B-aligned rows, staggered banks.
#define XS_STRIDE 392   // x window, bf16
#define QK_STRIDE 40    // q / k per head, bf16
#define VT_STRIDE 72    // v transposed [d][n], bf16
#define SS_STRIDE 68    // scores, f32
#define PS_STRIDE 72    // probs, bf16
#define AO_STRIDE 392   // attention output (all heads), bf16
#define WP_STRIDE 40    // staged qkv weight k-panel [96][32], bf16

// Packed conversions (backend: v_cvt_pk_bf16_f32).
__device__ __forceinline__ v4bf cvt4(v4f f) {
  return __builtin_convertvector(f, v4bf);
}
__device__ __forceinline__ v8bf cvt8(v8f f) {
  return __builtin_convertvector(f, v8bf);
}

// A-fragment (16x32 bf16, MxK): lane loads contiguous k (this code's order, same as B) = k0+half*16+[0..15].
// Two 16B LDS loads.
__device__ __forceinline__ v16bf load_a16(const __bf16* base, int row,
                                          int stride, int kbase) {
  const __bf16* p = base + row * stride + kbase;
  v8bf lo = *(const v8bf*)(p);
  v8bf hi = *(const v8bf*)(p + 8);
  return __builtin_shufflevector(lo, hi, 0,1,2,3,4,5,6,7,
                                         8,9,10,11,12,13,14,15);
}

// B-fragment (32x16 bf16, KxN): lane loads contiguous k (this code's order, same as A) = k0+half*16+[0..15].
__device__ __forceinline__ v16bf load_b16_lds(const __bf16* base, int row,
                                              int stride, int kbase) {
  const __bf16* p = base + row * stride + kbase;
  v8bf lo = *(const v8bf*)(p);
  v8bf hi = *(const v8bf*)(p + 8);
  return __builtin_shufflevector(lo, hi, 0,1,2,3,4,5,6,7,
                                         8,9,10,11,12,13,14,15);
}

// B-fragment straight from a row-major fp32 weight matrix (B = W^T).
__device__ __forceinline__ v16bf load_b16_w(const float* __restrict__ W,
                                            int wrow, int ldw, int kbase) {
  const v4f* p = (const v4f*)(W + (size_t)wrow * ldw + kbase);
  v4bf b0 = cvt4(p[0]);
  v4bf b1 = cvt4(p[1]);
  v4bf b2 = cvt4(p[2]);
  v4bf b3 = cvt4(p[3]);
  v8bf lo = __builtin_shufflevector(b0, b1, 0,1,2,3,4,5,6,7);
  v8bf hi = __builtin_shufflevector(b2, b3, 0,1,2,3,4,5,6,7);
  return __builtin_shufflevector(lo, hi, 0,1,2,3,4,5,6,7,
                                         8,9,10,11,12,13,14,15);
}

__device__ __forceinline__ v8f wmma_bf16(v16bf a, v16bf b, v8f c) {
  return __builtin_amdgcn_wmma_f32_16x16x32_bf16(false, a, false, b,
                                                 (short)0, c, false, false);
}

__global__ __launch_bounds__(256)
void win_attn_fused(const float* __restrict__ x,
                    const float* __restrict__ mask,
                    const float* __restrict__ qkv_w,
                    const float* __restrict__ qkv_b,
                    const float* __restrict__ proj_w,
                    const float* __restrict__ proj_b,
                    float* __restrict__ out) {
  __shared__ __attribute__((aligned(16))) __bf16 xs [NP * XS_STRIDE];   // 50.2KB
  __shared__ __attribute__((aligned(16))) __bf16 qs [NP * QK_STRIDE];   //  5.1KB
  __shared__ __attribute__((aligned(16))) __bf16 ks2[NP * QK_STRIDE];   //  5.1KB
  __shared__ __attribute__((aligned(16))) __bf16 vts[HD * VT_STRIDE];   //  4.6KB
  __shared__ __attribute__((aligned(16))) float  ss [NP * SS_STRIDE];   // 17.4KB
  __shared__ __attribute__((aligned(16))) __bf16 ps [NP * PS_STRIDE];   //  9.2KB
  __shared__ __attribute__((aligned(16))) __bf16 aos[NP * AO_STRIDE];   // 50.2KB
  __shared__ __attribute__((aligned(16))) __bf16 wp [2][96 * WP_STRIDE];// 15.4KB

  const int b    = blockIdx.x;
  const int tid  = threadIdx.x;
  const int wv   = tid >> 5;          // wave id, 0..7
  const int lane = tid & 31;
  const int half = (lane >> 4) & 1;
  const int n16  = lane & 15;
  const float scale = 0.17677669529663687f; // 1/sqrt(32)

  // ---- Phase 0: x window -> LDS bf16, pad rows zeroed --------------------
  for (int idx = tid; idx < NP * (CDIM / 4); idx += 256) {
    int row = idx / (CDIM / 4);
    int c4  = (idx % (CDIM / 4)) * 4;
    v4f f = {0.f, 0.f, 0.f, 0.f};
    if (row < SEQ)
      f = *(const v4f*)(x + ((size_t)b * SEQ + row) * CDIM + c4);
    *(v4bf*)(&xs[row * XS_STRIDE + c4]) = cvt4(f);
  }
  __syncthreads();

  // ---- Phase 1: per-head QKV -> attention -------------------------------
  for (int h = 0; h < NH; ++h) {
    // stage the 96x32 fp32 weight k-panel for step kk into bf16 LDS panel
    auto stage = [&](int kk, __bf16* dst) {
#pragma unroll
      for (int e = 0; e < 3; ++e) {
        int idx = tid + e * 256;            // 768 float4 chunks: 96 rows x 8
        int j   = idx >> 3;
        int c4  = (idx & 7) * 4;
        int wr  = (j >> 5) * CDIM + h * HD + (j & 31);
        v4f f = *(const v4f*)(qkv_w + (size_t)wr * CDIM + kk * 32 + c4);
        *(v4bf*)(dst + j * WP_STRIDE + c4) = cvt4(f);
      }
    };

    // 1a: q,k,v for this head: [64x384] @ [384x96].  Workgroup-scope k-loop,
    // double-buffered weight panel (stage kk+1 while computing kk),
    // one barrier per k-step.  Each wave owns 3 of the 24 output tiles.
    const int t0 = wv * 3;
    const int ct0 = (t0    ) >> 2, rt0 = (t0    ) & 3;
    const int ct1 = (t0 + 1) >> 2, rt1 = (t0 + 1) & 3;
    const int ct2 = (t0 + 2) >> 2, rt2 = (t0 + 2) & 3;
    v8f acc0 = {}, acc1 = {}, acc2 = {};

    stage(0, wp[0]);
    __syncthreads();
    for (int kk = 0; kk < 12; ++kk) {
      int cur = kk & 1;
      if (kk < 11) stage(kk + 1, wp[cur ^ 1]);

      int ab = kk * 32 + half * 16;
      v16bf a0 = load_a16(xs, rt0 * 16 + n16, XS_STRIDE, ab);
      v16bf a1 = load_a16(xs, rt1 * 16 + n16, XS_STRIDE, ab);
      v16bf a2 = load_a16(xs, rt2 * 16 + n16, XS_STRIDE, ab);
      v16bf b0 = load_b16_lds(wp[cur], ct0 * 16 + n16, WP_STRIDE, half * 16);
      v16bf b1 = load_b16_lds(wp[cur], ct1 * 16 + n16, WP_STRIDE, half * 16);
      v16bf b2 = load_b16_lds(wp[cur], ct2 * 16 + n16, WP_STRIDE, half * 16);
      acc0 = wmma_bf16(a0, b0, acc0);
      acc1 = wmma_bf16(a1, b1, acc1);
      acc2 = wmma_bf16(a2, b2, acc2);
      __syncthreads();
    }

    // epilogue: bias + packed bf16 convert + scatter to qs / ks / vts
#pragma unroll
    for (int i = 0; i < 3; ++i) {
      int ct = (i == 0) ? ct0 : (i == 1) ? ct1 : ct2;
      int rt = (i == 0) ? rt0 : (i == 1) ? rt1 : rt2;
      v8f acc = (i == 0) ? acc0 : (i == 1) ? acc1 : acc2;
      int g  = ct >> 1;
      int dt = (ct & 1) * 16;
      float bias = qkv_b[g * CDIM + h * HD + dt + n16];
      v8bf ab = cvt8(acc + bias);
#pragma unroll
      for (int r = 0; r < 8; ++r) {
        int row = rt * 16 + half * 8 + r;
        if (g == 0)      qs [row * QK_STRIDE + dt + n16]   = ab[r];
        else if (g == 1) ks2[row * QK_STRIDE + dt + n16]   = ab[r];
        else             vts[(dt + n16) * VT_STRIDE + row] = ab[r];  // V^T
      }
    }
    __syncthreads();

    // 1b: S = Q K^T * scale + mask.  16 tiles / 8 waves, single k-step.
    for (int i = 0; i < 2; ++i) {
      int t  = wv * 2 + i;
      int rt = t >> 2;
      int ck = t & 3;
      v8f acc = {};
      v16bf a  = load_a16(qs, rt * 16 + n16, QK_STRIDE, half * 16);
      v16bf bb = load_b16_lds(ks2, ck * 16 + n16, QK_STRIDE, half * 16);
      acc = wmma_bf16(a, bb, acc);
#pragma unroll
      for (int r = 0; r < 8; ++r) {
        int row = rt * 16 + half * 8 + r;
        int col = ck * 16 + n16;
        float sv = -1e30f;
        if (row < SEQ && col < SEQ)
          sv = acc[r] * scale + mask[((size_t)b * SEQ + row) * SEQ + col];
        ss[row * SS_STRIDE + col] = sv;
      }
    }
    __syncthreads();

    // 1c: masked softmax; 4 lanes per row, exp cached in registers,
    // probabilities emitted as packed 8-byte bf16 stores.
    {
      int row = tid >> 2;
      int sub = tid & 3;
      const float* sr = &ss[row * SS_STRIDE + sub * 16];
      float mx = -1e30f;
#pragma unroll
      for (int j = 0; j < 16; ++j) mx = fmaxf(mx, sr[j]);
      mx = fmaxf(mx, __shfl_xor(mx, 1));
      mx = fmaxf(mx, __shfl_xor(mx, 2));
      float ev[16];
      float sum = 0.f;
#pragma unroll
      for (int j = 0; j < 16; ++j) { ev[j] = __expf(sr[j] - mx); sum += ev[j]; }
      sum += __shfl_xor(sum, 1);
      sum += __shfl_xor(sum, 2);
      float inv = 1.0f / sum;
      __bf16* pr = &ps[row * PS_STRIDE + sub * 16];
#pragma unroll
      for (int q = 0; q < 4; ++q) {
        v4f t = {ev[q*4+0] * inv, ev[q*4+1] * inv,
                 ev[q*4+2] * inv, ev[q*4+3] * inv};
        *(v4bf*)(pr + q * 4) = cvt4(t);
      }
    }
    __syncthreads();

    // 1d: O = P V.  8 tiles / 8 waves, 2 k-steps over 64 keys.
    {
      int rt = wv >> 1;
      int cd = wv & 1;
      v8f acc = {};
#pragma unroll
      for (int kk = 0; kk < 2; ++kk) {
        v16bf a  = load_a16(ps, rt * 16 + n16, PS_STRIDE, kk * 32 + half * 16);
        v16bf bb = load_b16_lds(vts, cd * 16 + n16, VT_STRIDE,
                                kk * 32 + half * 16);
        acc = wmma_bf16(a, bb, acc);
      }
      v8bf ab = cvt8(acc);
#pragma unroll
      for (int r = 0; r < 8; ++r) {
        int row = rt * 16 + half * 8 + r;
        aos[row * AO_STRIDE + h * HD + cd * 16 + n16] = ab[r];
      }
    }
    __syncthreads();
  }

  // ---- Phase 2: out = AO @ proj_w^T + proj_b ----------------------------
  // Each wave owns 3 column tiles; B-fragment loaded/converted once and
  // reused across all 4 row tiles (independent WMMA chains hide hazards).
  for (int j = 0; j < 3; ++j) {
    int ct   = wv * 3 + j;
    int wrow = ct * 16 + n16;
    v8f a0 = {}, a1 = {}, a2 = {}, a3 = {};
    for (int kk = 0; kk < 12; ++kk) {
      v16bf bb = load_b16_w(proj_w, wrow, CDIM, kk * 32 + half * 16);
      int ab = kk * 32 + half * 16;
      a0 = wmma_bf16(load_a16(aos, 0 * 16 + n16, AO_STRIDE, ab), bb, a0);
      a1 = wmma_bf16(load_a16(aos, 1 * 16 + n16, AO_STRIDE, ab), bb, a1);
      a2 = wmma_bf16(load_a16(aos, 2 * 16 + n16, AO_STRIDE, ab), bb, a2);
      a3 = wmma_bf16(load_a16(aos, 3 * 16 + n16, AO_STRIDE, ab), bb, a3);
    }
    float bias = proj_b[wrow];
#pragma unroll
    for (int rt = 0; rt < 4; ++rt) {
      v8f acc = (rt == 0) ? a0 : (rt == 1) ? a1 : (rt == 2) ? a2 : a3;
#pragma unroll
      for (int r = 0; r < 8; ++r) {
        int row = rt * 16 + half * 8 + r;
        if (row < SEQ)
          out[((size_t)b * SEQ + row) * CDIM + ct * 16 + n16] = acc[r] + bias;
      }
    }
  }
}

extern "C" void kernel_launch(void* const* d_in, const int* in_sizes, int n_in,
                              void* d_out, int out_size, void* d_ws,
                              size_t ws_size, hipStream_t stream) {
  const float* x      = (const float*)d_in[0];
  const float* mask   = (const float*)d_in[1];
  const float* qkv_w  = (const float*)d_in[2];
  const float* qkv_b  = (const float*)d_in[3];
  const float* proj_w = (const float*)d_in[4];
  const float* proj_b = (const float*)d_in[5];
  float* out = (float*)d_out;

  const int nwin = in_sizes[0] / (SEQ * CDIM);   // 4096
  dim3 grid(nwin), block(256);
  win_attn_fused<<<grid, block, 0, stream>>>(x, mask, qkv_w, qkv_b,
                                             proj_w, proj_b, out);
}
